// MPNNProcessor_47425028883051
// MI455X (gfx1250) — compile-verified, matrix operand order checked by reading, not run on hardware
//
#include <hip/hip_runtime.h>

// MPNN processor for MI455X (gfx1250): bf16 WMMA GEMMs (2 M-tiles per wave,
// rotated B-fragment pipeline), fp32 accum, f32 atomic scatter-add for
// segment_sum, fp32 residual + LayerNorm.

typedef __bf16 bf16_t;
typedef __attribute__((ext_vector_type(16))) __bf16 v16bf;
typedef __attribute__((ext_vector_type(8)))  float  v8f;

#define Hdim 128
#define EDIM 32

// Strides chosen: (stride_in_dwords % 64) * row distinct per 16 rows -> no LDS
// bank conflicts; strides are multiples of 8 elements so uint4 loads align.
#define EAST 296   // edge A-tile stride (288 used)
#define ACTST 136  // activation tile stride (128 used)
#define NAST 264   // node A-tile stride (256 used)

__device__ __forceinline__ v16bf lds_frag(const bf16_t* p) {
  union { uint4 q[2]; v16bf v; } u;
  u.q[0] = *(const uint4*)p;
  u.q[1] = *(const uint4*)(p + 8);
  return u.v;
}

__device__ __forceinline__ v16bf load_b(const bf16_t* wp, int frag, int lane) {
  return *(const v16bf*)(wp + ((size_t)frag * 32 + lane) * 16);
}

#define WMMA_BF16(a, b, c) \
  __builtin_amdgcn_wmma_f32_16x16x32_bf16(false, (a), false, (b), (short)0, (c), false, false)

// ---------------------------------------------------------------------------
// Pack fp32 row-major W[K][N] into bf16 WMMA-B fragments.
// Fragment fi = kb*(N/16)+nt holds 512 bf16: per lane L (0..31), 16 contiguous
// values = W[kb*32 + (L>>4)*16 + i][nt*16 + (L&15)], i = 0..15.
// ---------------------------------------------------------------------------
__global__ void pack_w(const float* __restrict__ W, bf16_t* __restrict__ out,
                       int K, int N) {
  int o = blockIdx.x * blockDim.x + threadIdx.x;
  if (o >= K * N) return;
  int i    = o & 15;
  int lane = (o >> 4) & 31;
  int fi   = o >> 9;
  int NT   = N >> 4;
  int nt   = fi % NT;
  int kb   = fi / NT;
  int n    = nt * 16 + (lane & 15);
  int k    = kb * 32 + ((lane >> 4) << 4) + i;
  out[o] = (bf16_t)W[(size_t)k * N + n];
}

__global__ void f32_to_bf16(const float* __restrict__ x, bf16_t* __restrict__ y,
                            long long n) {
  long long i = (long long)blockIdx.x * blockDim.x + threadIdx.x;
  if (i < n) y[i] = (bf16_t)x[i];
}

__global__ void zero_f32(float4* __restrict__ p, long long n4) {
  long long i = (long long)blockIdx.x * blockDim.x + threadIdx.x;
  if (i < n4) { float4 z; z.x = z.y = z.z = z.w = 0.f; p[i] = z; }
}

// ---------------------------------------------------------------------------
// Edge kernel: 2 waves/block, each wave owns 32 edges (2 M-tiles).
// A = [h[src] | h[dst] | edge_attr] (K=288), 3-layer MLP
// (288->128 relu, 128->128 relu, 128->128), atomic scatter-add into agg[dst].
// ---------------------------------------------------------------------------
__global__ __launch_bounds__(64) void mpnn_edge_kernel(
    const bf16_t* __restrict__ hb, const float* __restrict__ ea,
    const long long* __restrict__ srcI, const long long* __restrict__ dstI,
    const bf16_t* __restrict__ w1p, const float* __restrict__ b1,
    const bf16_t* __restrict__ w2p, const float* __restrict__ b2,
    const bf16_t* __restrict__ w3p, const float* __restrict__ b3,
    float* __restrict__ agg, int nEdges) {
  __shared__ alignas(32) bf16_t atile[2][32][EAST];
  __shared__ alignas(32) bf16_t act[2][32][ACTST];
  __shared__ long long sidx[2][32];
  __shared__ long long didx[2][32];

  const int lane = threadIdx.x & 31;
  const int wave = threadIdx.x >> 5;
  const int base = (blockIdx.x * 2 + wave) * 32;
  if (base >= nEdges) return;

  __builtin_prefetch(w2p, 0, 1);
  __builtin_prefetch(w3p, 0, 1);

  // Stage all 32 src/dst indices with one coalesced load per array.
  {
    long long ei = base + lane;
    if (ei >= nEdges) ei = nEdges - 1;
    sidx[wave][lane] = srcI[ei];
    didx[wave][lane] = dstI[ei];
  }

  // Gather bf16 A tile: cols [0,128)=h[src], [128,256)=h[dst], [256,288)=attr
  for (int e = 0; e < 32; ++e) {
    long long edge = base + e;
    if (edge >= nEdges) edge = nEdges - 1;
    const long long s = sidx[wave][e];
    const long long d = didx[wave][e];
    const uint2* ps = (const uint2*)(hb + s * Hdim);
    const uint2* pd = (const uint2*)(hb + d * Hdim);
    *(uint2*)&atile[wave][e][lane * 4]        = ps[lane];
    *(uint2*)&atile[wave][e][Hdim + lane * 4] = pd[lane];
    atile[wave][e][2 * Hdim + lane] = (bf16_t)ea[edge * EDIM + lane];
  }

  const int row   = lane & 15;
  const int khalf = (lane >> 4) << 4;  // K offset 0/16 in a 32-K block
  const int ncol  = lane & 15;
  const int mhalf = (lane >> 4) << 3;  // C rows: lanes16-31 hold M+8

  const bf16_t* aw = &atile[wave][0][0];
  bf16_t*       cw = &act[wave][0][0];

  v8f zero = {};
  v8f acc0[8], acc1[8];

  // ---- layer 1: K=288 (9 k-blocks) ----
#pragma unroll
  for (int nt = 0; nt < 8; ++nt) { acc0[nt] = zero; acc1[nt] = zero; }
#pragma unroll
  for (int kb = 0; kb < 9; ++kb) {
    v16bf a0 = lds_frag(aw + (size_t)row * EAST + kb * 32 + khalf);
    v16bf a1 = lds_frag(aw + (size_t)(row + 16) * EAST + kb * 32 + khalf);
    v16bf b = load_b(w1p, kb * 8, lane);
#pragma unroll
    for (int nt = 0; nt < 8; ++nt) {
      v16bf bn = b;
      if (nt < 7) bn = load_b(w1p, kb * 8 + nt + 1, lane);
      acc0[nt] = WMMA_BF16(a0, b, acc0[nt]);
      acc1[nt] = WMMA_BF16(a1, b, acc1[nt]);
      b = bn;
    }
  }
#pragma unroll
  for (int nt = 0; nt < 8; ++nt) {
    float bias = b1[nt * 16 + ncol];
#pragma unroll
    for (int r = 0; r < 8; ++r) {
      float v0 = acc0[nt][r] + bias;
      float v1 = acc1[nt][r] + bias;
      cw[(size_t)(r + mhalf) * ACTST + nt * 16 + ncol]      = (bf16_t)(v0 > 0.f ? v0 : 0.f);
      cw[(size_t)(16 + r + mhalf) * ACTST + nt * 16 + ncol] = (bf16_t)(v1 > 0.f ? v1 : 0.f);
    }
  }

  // ---- layer 2: K=128 (4 k-blocks) ----
#pragma unroll
  for (int nt = 0; nt < 8; ++nt) { acc0[nt] = zero; acc1[nt] = zero; }
#pragma unroll
  for (int kb = 0; kb < 4; ++kb) {
    v16bf a0 = lds_frag(cw + (size_t)row * ACTST + kb * 32 + khalf);
    v16bf a1 = lds_frag(cw + (size_t)(row + 16) * ACTST + kb * 32 + khalf);
    v16bf b = load_b(w2p, kb * 8, lane);
#pragma unroll
    for (int nt = 0; nt < 8; ++nt) {
      v16bf bn = b;
      if (nt < 7) bn = load_b(w2p, kb * 8 + nt + 1, lane);
      acc0[nt] = WMMA_BF16(a0, b, acc0[nt]);
      acc1[nt] = WMMA_BF16(a1, b, acc1[nt]);
      b = bn;
    }
  }
#pragma unroll
  for (int nt = 0; nt < 8; ++nt) {
    float bias = b2[nt * 16 + ncol];
#pragma unroll
    for (int r = 0; r < 8; ++r) {
      float v0 = acc0[nt][r] + bias;
      float v1 = acc1[nt][r] + bias;
      cw[(size_t)(r + mhalf) * ACTST + nt * 16 + ncol]      = (bf16_t)(v0 > 0.f ? v0 : 0.f);
      cw[(size_t)(16 + r + mhalf) * ACTST + nt * 16 + ncol] = (bf16_t)(v1 > 0.f ? v1 : 0.f);
    }
  }

  // ---- layer 3: K=128, then atomic scatter-add into agg[dst] ----
#pragma unroll
  for (int nt = 0; nt < 8; ++nt) { acc0[nt] = zero; acc1[nt] = zero; }
#pragma unroll
  for (int kb = 0; kb < 4; ++kb) {
    v16bf a0 = lds_frag(cw + (size_t)row * ACTST + kb * 32 + khalf);
    v16bf a1 = lds_frag(cw + (size_t)(row + 16) * ACTST + kb * 32 + khalf);
    v16bf b = load_b(w3p, kb * 8, lane);
#pragma unroll
    for (int nt = 0; nt < 8; ++nt) {
      v16bf bn = b;
      if (nt < 7) bn = load_b(w3p, kb * 8 + nt + 1, lane);
      acc0[nt] = WMMA_BF16(a0, b, acc0[nt]);
      acc1[nt] = WMMA_BF16(a1, b, acc1[nt]);
      b = bn;
    }
  }
#pragma unroll
  for (int nt = 0; nt < 8; ++nt) {
    float bias = b3[nt * 16 + ncol];
#pragma unroll
    for (int r = 0; r < 8; ++r) {
      const int e0 = r + mhalf;
      const int e1 = 16 + r + mhalf;
      if (base + e0 < nEdges) {
        const long long d = didx[wave][e0];
        unsafeAtomicAdd(&agg[d * Hdim + nt * 16 + ncol], acc0[nt][r] + bias);
      }
      if (base + e1 < nEdges) {
        const long long d = didx[wave][e1];
        unsafeAtomicAdd(&agg[d * Hdim + nt * 16 + ncol], acc1[nt][r] + bias);
      }
    }
  }
}

// ---------------------------------------------------------------------------
// Node kernel: 2 waves/block, each wave owns 32 nodes (2 M-tiles).
// A = [h | agg] (K=256), 3-layer MLP, residual + LayerNorm, writes fp32 h_out
// and a bf16 copy for the next step's gathers.
// ---------------------------------------------------------------------------
__global__ __launch_bounds__(64) void mpnn_node_kernel(
    const float* __restrict__ hin, const bf16_t* __restrict__ hbin,
    const float* __restrict__ agg,
    const bf16_t* __restrict__ w1p, const float* __restrict__ b1,
    const bf16_t* __restrict__ w2p, const float* __restrict__ b2,
    const bf16_t* __restrict__ w3p, const float* __restrict__ b3,
    const float* __restrict__ ln_g, const float* __restrict__ ln_b,
    float* __restrict__ hout, bf16_t* __restrict__ hbout, int nNodes) {
  __shared__ alignas(32) bf16_t atile[2][32][NAST];
  __shared__ alignas(32) bf16_t act[2][32][ACTST];
  __shared__ float sg[Hdim], sb[Hdim];

  if (threadIdx.x < 64) {
    sg[threadIdx.x] = ln_g[threadIdx.x];
    sg[threadIdx.x + 64] = ln_g[threadIdx.x + 64];
    sb[threadIdx.x] = ln_b[threadIdx.x];
    sb[threadIdx.x + 64] = ln_b[threadIdx.x + 64];
  }
  __syncthreads();

  const int lane = threadIdx.x & 31;
  const int wave = threadIdx.x >> 5;
  const int base = (blockIdx.x * 2 + wave) * 32;
  if (base >= nNodes) return;

  // Gather A tile: cols [0,128) = h (bf16), [128,256) = agg (f32 -> bf16)
  for (int e = 0; e < 32; ++e) {
    int node = base + e;
    if (node >= nNodes) node = nNodes - 1;
    const uint2* ph = (const uint2*)(hbin + (size_t)node * Hdim);
    *(uint2*)&atile[wave][e][lane * 4] = ph[lane];
    const float4* pa = (const float4*)(agg + (size_t)node * Hdim);
    float4 v = pa[lane];
    bf16_t* o = &atile[wave][e][Hdim + lane * 4];
    o[0] = (bf16_t)v.x; o[1] = (bf16_t)v.y; o[2] = (bf16_t)v.z; o[3] = (bf16_t)v.w;
  }

  const int row   = lane & 15;
  const int khalf = (lane >> 4) << 4;
  const int ncol  = lane & 15;
  const int mhalf = (lane >> 4) << 3;

  const bf16_t* aw = &atile[wave][0][0];
  bf16_t*       cw = &act[wave][0][0];

  v8f zero = {};
  v8f acc0[8], acc1[8];

  // ---- layer 1: K=256 (8 k-blocks) ----
#pragma unroll
  for (int nt = 0; nt < 8; ++nt) { acc0[nt] = zero; acc1[nt] = zero; }
#pragma unroll
  for (int kb = 0; kb < 8; ++kb) {
    v16bf a0 = lds_frag(aw + (size_t)row * NAST + kb * 32 + khalf);
    v16bf a1 = lds_frag(aw + (size_t)(row + 16) * NAST + kb * 32 + khalf);
    v16bf b = load_b(w1p, kb * 8, lane);
#pragma unroll
    for (int nt = 0; nt < 8; ++nt) {
      v16bf bn = b;
      if (nt < 7) bn = load_b(w1p, kb * 8 + nt + 1, lane);
      acc0[nt] = WMMA_BF16(a0, b, acc0[nt]);
      acc1[nt] = WMMA_BF16(a1, b, acc1[nt]);
      b = bn;
    }
  }
#pragma unroll
  for (int nt = 0; nt < 8; ++nt) {
    float bias = b1[nt * 16 + ncol];
#pragma unroll
    for (int r = 0; r < 8; ++r) {
      float v0 = acc0[nt][r] + bias;
      float v1 = acc1[nt][r] + bias;
      cw[(size_t)(r + mhalf) * ACTST + nt * 16 + ncol]      = (bf16_t)(v0 > 0.f ? v0 : 0.f);
      cw[(size_t)(16 + r + mhalf) * ACTST + nt * 16 + ncol] = (bf16_t)(v1 > 0.f ? v1 : 0.f);
    }
  }

  // ---- layer 2 ----
#pragma unroll
  for (int nt = 0; nt < 8; ++nt) { acc0[nt] = zero; acc1[nt] = zero; }
#pragma unroll
  for (int kb = 0; kb < 4; ++kb) {
    v16bf a0 = lds_frag(cw + (size_t)row * ACTST + kb * 32 + khalf);
    v16bf a1 = lds_frag(cw + (size_t)(row + 16) * ACTST + kb * 32 + khalf);
    v16bf b = load_b(w2p, kb * 8, lane);
#pragma unroll
    for (int nt = 0; nt < 8; ++nt) {
      v16bf bn = b;
      if (nt < 7) bn = load_b(w2p, kb * 8 + nt + 1, lane);
      acc0[nt] = WMMA_BF16(a0, b, acc0[nt]);
      acc1[nt] = WMMA_BF16(a1, b, acc1[nt]);
      b = bn;
    }
  }
#pragma unroll
  for (int nt = 0; nt < 8; ++nt) {
    float bias = b2[nt * 16 + ncol];
#pragma unroll
    for (int r = 0; r < 8; ++r) {
      float v0 = acc0[nt][r] + bias;
      float v1 = acc1[nt][r] + bias;
      cw[(size_t)(r + mhalf) * ACTST + nt * 16 + ncol]      = (bf16_t)(v0 > 0.f ? v0 : 0.f);
      cw[(size_t)(16 + r + mhalf) * ACTST + nt * 16 + ncol] = (bf16_t)(v1 > 0.f ? v1 : 0.f);
    }
  }

  // ---- layer 3 ----
#pragma unroll
  for (int nt = 0; nt < 8; ++nt) { acc0[nt] = zero; acc1[nt] = zero; }
#pragma unroll
  for (int kb = 0; kb < 4; ++kb) {
    v16bf a0 = lds_frag(cw + (size_t)row * ACTST + kb * 32 + khalf);
    v16bf a1 = lds_frag(cw + (size_t)(row + 16) * ACTST + kb * 32 + khalf);
    v16bf b = load_b(w3p, kb * 8, lane);
#pragma unroll
    for (int nt = 0; nt < 8; ++nt) {
      v16bf bn = b;
      if (nt < 7) bn = load_b(w3p, kb * 8 + nt + 1, lane);
      acc0[nt] = WMMA_BF16(a0, b, acc0[nt]);
      acc1[nt] = WMMA_BF16(a1, b, acc1[nt]);
      b = bn;
    }
  }

  // upd tile (fp32) into LDS: reuse this wave's atile chunk (32*NAST*2 bytes
  // = 16896 = exactly 32 rows * 132 floats).
  float* updt = (float*)&atile[wave][0][0];
#pragma unroll
  for (int nt = 0; nt < 8; ++nt) {
    float bias = b3[nt * 16 + ncol];
#pragma unroll
    for (int r = 0; r < 8; ++r) {
      updt[(size_t)(r + mhalf) * 132 + nt * 16 + ncol]      = acc0[nt][r] + bias;
      updt[(size_t)(16 + r + mhalf) * 132 + nt * 16 + ncol] = acc1[nt][r] + bias;
    }
  }

  // residual + LayerNorm: each of the 32 lanes owns one node row
  const int node = base + lane;
  if (node < nNodes) {
    const float* hrow = hin + (size_t)node * Hdim;
    const float* urow = updt + (size_t)lane * 132;
    float s1 = 0.f, s2 = 0.f;
    for (int f = 0; f < Hdim; ++f) {
      float x = hrow[f] + urow[f];
      s1 += x; s2 += x * x;
    }
    const float mu  = s1 * (1.f / Hdim);
    const float var = s2 * (1.f / Hdim) - mu * mu;
    const float inv = rsqrtf(var + 1e-5f);
    float*  orow = hout  + (size_t)node * Hdim;
    bf16_t* brow = hbout + (size_t)node * Hdim;
    for (int f = 0; f < Hdim; ++f) {
      float x = hrow[f] + urow[f];
      float y = (x - mu) * inv * sg[f] + sb[f];
      orow[f] = y;
      brow[f] = (bf16_t)y;
    }
  }
}

// ---------------------------------------------------------------------------
extern "C" void kernel_launch(void* const* d_in, const int* in_sizes, int n_in,
                              void* d_out, int out_size, void* d_ws, size_t ws_size,
                              hipStream_t stream) {
  const float* h0  = (const float*)d_in[0];
  const float* ea  = (const float*)d_in[1];
  const float* mw1 = (const float*)d_in[2];
  const float* mb1 = (const float*)d_in[3];
  const float* mw2 = (const float*)d_in[4];
  const float* mb2 = (const float*)d_in[5];
  const float* mw3 = (const float*)d_in[6];
  const float* mb3 = (const float*)d_in[7];
  const float* uw1 = (const float*)d_in[8];
  const float* ub1 = (const float*)d_in[9];
  const float* uw2 = (const float*)d_in[10];
  const float* ub2 = (const float*)d_in[11];
  const float* uw3 = (const float*)d_in[12];
  const float* ub3 = (const float*)d_in[13];
  const float* lng = (const float*)d_in[14];
  const float* lnb = (const float*)d_in[15];
  const long long* eidx = (const long long*)d_in[16];

  const int Nn = in_sizes[0] / Hdim;       // 50000
  const int E  = in_sizes[1] / EDIM;       // 800000
  const long long* srcI = eidx;
  const long long* dstI = eidx + E;

  char* ws = (char*)d_ws;
  size_t off = 0;
  auto carve = [&](size_t bytes) -> char* {
    char* p = ws + off;
    off = (off + bytes + 255) & ~(size_t)255;
    return p;
  };

  float*  hB   = (float*)carve((size_t)Nn * Hdim * 4);
  float*  hA   = (float*)carve((size_t)Nn * Hdim * 4);
  bf16_t* bA   = (bf16_t*)carve((size_t)Nn * Hdim * 2);
  bf16_t* bB   = (bf16_t*)carve((size_t)Nn * Hdim * 2);
  float*  agg  = (float*)carve((size_t)Nn * Hdim * 4);
  bf16_t* pmw1 = (bf16_t*)carve((size_t)(2 * Hdim + EDIM) * Hdim * 2);
  bf16_t* pmw2 = (bf16_t*)carve((size_t)Hdim * Hdim * 2);
  bf16_t* pmw3 = (bf16_t*)carve((size_t)Hdim * Hdim * 2);
  bf16_t* puw1 = (bf16_t*)carve((size_t)(2 * Hdim) * Hdim * 2);
  bf16_t* puw2 = (bf16_t*)carve((size_t)Hdim * Hdim * 2);
  bf16_t* puw3 = (bf16_t*)carve((size_t)Hdim * Hdim * 2);

  auto pack = [&](const float* W, bf16_t* P, int K, int Ncols) {
    int total = K * Ncols;
    pack_w<<<(total + 255) / 256, 256, 0, stream>>>(W, P, K, Ncols);
  };
  pack(mw1, pmw1, 2 * Hdim + EDIM, Hdim);
  pack(mw2, pmw2, Hdim, Hdim);
  pack(mw3, pmw3, Hdim, Hdim);
  pack(uw1, puw1, 2 * Hdim, Hdim);
  pack(uw2, puw2, Hdim, Hdim);
  pack(uw3, puw3, Hdim, Hdim);

  const long long nh = (long long)Nn * Hdim;
  f32_to_bf16<<<(int)((nh + 255) / 256), 256, 0, stream>>>(h0, bA, nh);

  const int edgeBlocks = (E + 63) / 64;   // 64 edges per 2-wave block
  const int nodeBlocks = (Nn + 63) / 64;  // 64 nodes per 2-wave block
  const long long n4 = nh / 4;
  const int zeroBlocks = (int)((n4 + 255) / 256);

  const float*  hin  = h0;
  const bf16_t* hbin = bA;
  for (int s = 0; s < 3; ++s) {
    zero_f32<<<zeroBlocks, 256, 0, stream>>>((float4*)agg, n4);
    mpnn_edge_kernel<<<edgeBlocks, 64, 0, stream>>>(
        hbin, ea, srcI, dstI, pmw1, mb1, pmw2, mb2, pmw3, mb3, agg, E);
    float*  hout  = (s == 2) ? (float*)d_out : (s == 0 ? hB : hA);
    bf16_t* hbout = (s == 1) ? bA : bB;
    mpnn_node_kernel<<<nodeBlocks, 64, 0, stream>>>(
        hin, hbin, agg, puw1, ub1, puw2, ub2, puw3, ub3, lng, lnb,
        hout, hbout, Nn);
    hin = hout;
    hbin = hbout;
  }
}
